// GemmaAttention_63393717289746
// MI455X (gfx1250) — matrix operand order repaired and checked by reading, not compiled, not run on hardware
//
#include <hip/hip_runtime.h>
#include <hip/hip_bf16.h>

typedef __attribute__((ext_vector_type(16))) __bf16 v16bf;
typedef __attribute__((ext_vector_type(8)))  __bf16 v8bf;
typedef __attribute__((ext_vector_type(8)))  float  v8f;

#define S_LEN  2048
#define HID    2048
#define NHEAD  8
#define DHEAD  256

static constexpr float kScale = 0.0625f;       // 256^-0.5
static constexpr float kNeg   = -1.0e9f;

// ---------------------------------------------------------------------------
// gfx1250 async global->LDS copy (ASYNCcnt-tracked, no VGPR round trip)
// ---------------------------------------------------------------------------
__device__ __forceinline__ void async_copy_b128(const void* g, void* l) {
    // generic->LDS pointer: flat addr[31:0] is the LDS byte offset
    unsigned loff = (unsigned)(uintptr_t)l;
    asm volatile("global_load_async_to_lds_b128 %0, %1, off"
                 :: "v"(loff), "v"(g) : "memory");
}

template<int N>
__device__ __forceinline__ void wait_async() {
    asm volatile("s_wait_asynccnt %0" :: "n"(N));
}

// ---------------------------------------------------------------------------
// fp32 -> bf16 bulk convert (row-major copy)
// ---------------------------------------------------------------------------
__global__ __launch_bounds__(256)
void cvt_f32_bf16(const float* __restrict__ src, __bf16* __restrict__ dst, long n) {
    long i = ((long)blockIdx.x * blockDim.x + threadIdx.x) * 4;
    if (i >= n) return;
    float4 f = *(const float4*)(src + i);
    dst[i + 0] = (__bf16)f.x;
    dst[i + 1] = (__bf16)f.y;
    dst[i + 2] = (__bf16)f.z;
    dst[i + 3] = (__bf16)f.w;
}

// fp32 [K,N] -> bf16 transposed [N,K]  (coalesced reads, scattered writes)
__global__ __launch_bounds__(256)
void cvt_f32_bf16_T(const float* __restrict__ src, __bf16* __restrict__ dst,
                    int K, int N) {
    long idx = (long)blockIdx.x * blockDim.x + threadIdx.x;
    if (idx >= (long)K * N) return;
    int k = (int)(idx / N);
    int n = (int)(idx % N);
    dst[(size_t)n * K + k] = (__bf16)src[idx];
}

// ---------------------------------------------------------------------------
// Tiled WMMA GEMM:  C[M,N] = A[M,K] * B[K,N], B supplied as B^T [N][K] row-major.
//   A_F32     : A source fp32 (converted to bf16 while staging, sync path)
//   OUT_F32   : C written fp32, else bf16
//   OUT_TRANS : store C^T (C[cidx*ldc + row])
//   MASK      : epilogue c = c*scale + (col<=row ? 0 : NEG); fully-masked
//               tiles skip the whole K loop.
// Block: 256 threads (8 waves), 128x128 tile; wave tile 32x64 (2x4 WMMA).
// Double-buffered LDS + async copies: stage(next) overlaps compute(cur).
// ---------------------------------------------------------------------------
template<bool A_F32, bool OUT_F32, bool OUT_TRANS, bool MASK>
__global__ __launch_bounds__(256)
void wmma_gemm(const void* __restrict__ Av, long aBatch, int lda,
               const __bf16* __restrict__ Bt0, long bBatch, int ldb,
               void* __restrict__ Cv, long cBatch, int ldc,
               int K, float scale) {
    constexpr int BM = 128, BN = 128, BK = 32;
    constexpr int INFLIGHT = A_F32 ? 2 : 4;     // async ops issued per stage

    const int tid  = threadIdx.x;
    const int wave = tid >> 5;
    const int lane = tid & 31;
    const int l    = lane & 15;
    const int hi   = lane >> 4;
    const int wm   = (wave >> 1) * 32;          // 4 waves along M
    const int wn   = (wave & 1)  * 64;          // 2 waves along N
    const int m0   = blockIdx.y * BM;
    const int n0   = blockIdx.x * BN;

    const float*  Af = (const float*)Av  + (size_t)blockIdx.z * aBatch;
    const __bf16* Ab = (const __bf16*)Av + (size_t)blockIdx.z * aBatch;
    const __bf16* Bt = Bt0               + (size_t)blockIdx.z * bBatch;
    float*        Cf = (float*)Cv        + (size_t)blockIdx.z * cBatch;
    __bf16*       Cb = (__bf16*)Cv       + (size_t)blockIdx.z * cBatch;

    // +16B row pad: rows stay 16B-aligned, banks spread
    __shared__ __align__(16) __bf16 sA[2][128][40];
    __shared__ __align__(16) __bf16 sB[2][128][40];

    v8f acc[2][4];
#pragma unroll
    for (int i = 0; i < 2; ++i)
#pragma unroll
        for (int j = 0; j < 4; ++j) acc[i][j] = v8f{};

    // causal early-out: tile entirely above the diagonal -> pure NEG
    const int nK = (MASK && n0 > m0 + BM - 1) ? 0 : (K / BK);

    auto stage = [&](int buf, int k0) {
#pragma unroll
        for (int t = tid; t < 512; t += 256) {      // 128 rows x 4 chunks of 8
            const int r  = t >> 2;
            const int c8 = (t & 3) << 3;
            if constexpr (A_F32) {
                const float* src = Af + (size_t)(m0 + r) * lda + k0 + c8;
                float4 f0 = *(const float4*)src;
                float4 f1 = *(const float4*)(src + 4);
                __bf16* dst = &sA[buf][r][c8];
                dst[0] = (__bf16)f0.x; dst[1] = (__bf16)f0.y;
                dst[2] = (__bf16)f0.z; dst[3] = (__bf16)f0.w;
                dst[4] = (__bf16)f1.x; dst[5] = (__bf16)f1.y;
                dst[6] = (__bf16)f1.z; dst[7] = (__bf16)f1.w;
            } else {
                async_copy_b128(Ab + (size_t)(m0 + r) * lda + k0 + c8,
                                &sA[buf][r][c8]);
            }
            async_copy_b128(Bt + (size_t)(n0 + r) * ldb + k0 + c8,
                            &sB[buf][r][c8]);
        }
    };

    if (nK > 0) {
        stage(0, 0);
        for (int kt = 0; kt < nK; ++kt) {
            const int cur = kt & 1;
            if (kt + 1 < nK) {
                stage(cur ^ 1, (kt + 1) * BK);   // overlap with compute(cur)
                wait_async<INFLIGHT>();          // in-order: cur's copies done
            } else {
                wait_async<0>();
            }
            __syncthreads();

            // fragments per ISA 7.12.2 (wave32 bf16 16x16x32)
            v16bf afr[2], bfr[4];
#pragma unroll
            for (int i = 0; i < 2; ++i) {
                const __bf16* arow = &sA[cur][wm + i * 16 + l][0];
                v8bf alo = *(const v8bf*)(arow + hi * 8);       // K=hi*8..
                v8bf ahi = *(const v8bf*)(arow + 16 + hi * 8);  // K=16+hi*8..
                afr[i] = __builtin_shufflevector(alo, ahi,
                    0, 1, 2, 3, 4, 5, 6, 7, 8, 9, 10, 11, 12, 13, 14, 15);
            }
#pragma unroll
            for (int j = 0; j < 4; ++j) {
                const __bf16* brow = &sB[cur][wn + j * 16 + l][0];
                v8bf blo = *(const v8bf*)(brow + hi * 8);       // K=hi*8..
                v8bf bhi = *(const v8bf*)(brow + 16 + hi * 8);  // K=16+hi*8..
                bfr[j] = __builtin_shufflevector(blo, bhi,
                    0, 1, 2, 3, 4, 5, 6, 7, 8, 9, 10, 11, 12, 13, 14, 15);
            }
#pragma unroll
            for (int i = 0; i < 2; ++i)
#pragma unroll
                for (int j = 0; j < 4; ++j)
                    acc[i][j] = __builtin_amdgcn_wmma_f32_16x16x32_bf16(
                        false, afr[i], false, bfr[j], (short)0, acc[i][j],
                        false, false);
            __syncthreads();
        }
    }

    // epilogue (C/D layout: n = lane&15, m = (lane>>4)*8 + vgpr index)
#pragma unroll
    for (int i = 0; i < 2; ++i) {
#pragma unroll
        for (int j = 0; j < 4; ++j) {
            const int rbase = m0 + wm + i * 16 + hi * 8;
            const int cidx  = n0 + wn + j * 16 + l;
#pragma unroll
            for (int r = 0; r < 8; ++r) {
                float v = acc[i][j][r];
                const int row = rbase + r;
                if constexpr (MASK)
                    v = v * scale + ((cidx <= row) ? 0.0f : kNeg);
                const size_t idx = OUT_TRANS ? (size_t)cidx * ldc + row
                                             : (size_t)row  * ldc + cidx;
                if constexpr (OUT_F32) Cf[idx] = v;
                else                   Cb[idx] = (__bf16)v;
            }
        }
    }
}

// ---------------------------------------------------------------------------
// RoPE in place on bf16 Q [S, NH*DH] and K [S, DH]
// ---------------------------------------------------------------------------
__global__ __launch_bounds__(128)
void rope_kernel(__bf16* __restrict__ q, __bf16* __restrict__ k,
                 const int* __restrict__ pos_ids) {
    const int s = blockIdx.x;
    const int d = threadIdx.x;                   // 0..127 (half of DH)
    const float pos  = (float)pos_ids[s];
    const float invf = __expf(-(2.0f * (float)d / (float)DHEAD) * 9.210340371976184f);
    const float ang  = pos * invf;
    const float c  = __cosf(ang);
    const float sn = __sinf(ang);
#pragma unroll
    for (int h = 0; h < NHEAD; ++h) {
        __bf16* base = q + (size_t)s * HID + h * DHEAD;
        const float x1 = (float)base[d];
        const float x2 = (float)base[d + 128];
        base[d]       = (__bf16)(x1 * c - x2 * sn);
        base[d + 128] = (__bf16)(x2 * c + x1 * sn);
    }
    __bf16* kb = k + (size_t)s * DHEAD;
    const float x1 = (float)kb[d];
    const float x2 = (float)kb[d + 128];
    kb[d]       = (__bf16)(x1 * c - x2 * sn);
    kb[d + 128] = (__bf16)(x2 * c + x1 * sn);
}

// ---------------------------------------------------------------------------
// Row-wise softmax in place, row length = S_LEN, 256 threads x 8 elements
// ---------------------------------------------------------------------------
__global__ __launch_bounds__(256)
void softmax_kernel(float* __restrict__ attn) {
    const int tid  = threadIdx.x;
    const int wave = tid >> 5;
    const int lane = tid & 31;
    float* row = attn + (size_t)blockIdx.x * S_LEN;

    float4 a = ((const float4*)row)[2 * tid];
    float4 b = ((const float4*)row)[2 * tid + 1];

    float m = fmaxf(fmaxf(fmaxf(a.x, a.y), fmaxf(a.z, a.w)),
                    fmaxf(fmaxf(b.x, b.y), fmaxf(b.z, b.w)));
#pragma unroll
    for (int off = 16; off > 0; off >>= 1) m = fmaxf(m, __shfl_xor(m, off, 32));
    __shared__ float redm[8];
    if (lane == 0) redm[wave] = m;
    __syncthreads();
    float mm = redm[0];
#pragma unroll
    for (int w = 1; w < 8; ++w) mm = fmaxf(mm, redm[w]);

    a.x = __expf(a.x - mm); a.y = __expf(a.y - mm);
    a.z = __expf(a.z - mm); a.w = __expf(a.w - mm);
    b.x = __expf(b.x - mm); b.y = __expf(b.y - mm);
    b.z = __expf(b.z - mm); b.w = __expf(b.w - mm);

    float s = a.x + a.y + a.z + a.w + b.x + b.y + b.z + b.w;
#pragma unroll
    for (int off = 16; off > 0; off >>= 1) s += __shfl_xor(s, off, 32);
    __shared__ float reds[8];
    if (lane == 0) reds[wave] = s;
    __syncthreads();
    float tot = reds[0] + reds[1] + reds[2] + reds[3] +
                reds[4] + reds[5] + reds[6] + reds[7];
    const float inv = 1.0f / tot;
    a.x *= inv; a.y *= inv; a.z *= inv; a.w *= inv;
    b.x *= inv; b.y *= inv; b.z *= inv; b.w *= inv;
    ((float4*)row)[2 * tid]     = a;
    ((float4*)row)[2 * tid + 1] = b;
}

// ---------------------------------------------------------------------------
extern "C" void kernel_launch(void* const* d_in, const int* in_sizes, int n_in,
                              void* d_out, int out_size, void* d_ws, size_t ws_size,
                              hipStream_t stream) {
    const float* x    = (const float*)d_in[0];
    const int*   pos  = (const int*)  d_in[1];
    // d_in[2] attention_mask unused: causal mask generated in epilogue
    const float* w_q  = (const float*)d_in[3];
    const float* w_k  = (const float*)d_in[4];
    const float* w_v  = (const float*)d_in[5];
    const float* w_o  = (const float*)d_in[6];

    float* outp  = (float*)d_out;                       // [S, H]
    float* attnp = outp + (size_t)S_LEN * HID;          // [NH, S, S]

    // workspace layout (bf16); weights stored TRANSPOSED -> every GEMM gets
    // the pure-copy B^T staging path
    char* w = (char*)d_ws;
    __bf16* xb   = (__bf16*)w; w += (size_t)S_LEN * HID   * 2;
    __bf16* wqbT = (__bf16*)w; w += (size_t)HID   * HID   * 2;   // [2048][2048]
    __bf16* wkbT = (__bf16*)w; w += (size_t)DHEAD * HID   * 2;   // [256][2048]
    __bf16* wvbT = (__bf16*)w; w += (size_t)DHEAD * HID   * 2;   // [256][2048]
    __bf16* wobT = (__bf16*)w; w += (size_t)HID   * HID   * 2;   // [2048][2048]
    __bf16* qb   = (__bf16*)w; w += (size_t)S_LEN * HID   * 2;   // [S, NH*DH]
    __bf16* kb   = (__bf16*)w; w += (size_t)S_LEN * DHEAD * 2;   // [S, DH]
    __bf16* vbT  = (__bf16*)w; w += (size_t)DHEAD * S_LEN * 2;   // [DH, S] (V^T)
    __bf16* ctx  = (__bf16*)w; w += (size_t)S_LEN * HID   * 2;   // [S, NH*DH]

    cvt_f32_bf16<<<dim3((unsigned)(((long)S_LEN * HID / 4 + 255) / 256)),
                   256, 0, stream>>>(x, xb, (long)S_LEN * HID);
    auto cvtT = [&](const float* s, __bf16* d, int K, int N) {
        long n = (long)K * N;
        cvt_f32_bf16_T<<<dim3((unsigned)((n + 255) / 256)), 256, 0, stream>>>(s, d, K, N);
    };
    cvtT(w_q, wqbT, HID, HID);
    cvtT(w_k, wkbT, HID, DHEAD);
    cvtT(w_v, wvbT, HID, DHEAD);
    cvtT(w_o, wobT, HID, HID);

    // Q = X @ Wq  [2048 x 2048]
    wmma_gemm<false, false, false, false>
        <<<dim3(HID / 128, S_LEN / 128, 1), 256, 0, stream>>>(
        xb, 0, HID, wqbT, 0, HID, qb, 0, HID, HID, 1.0f);
    // K = X @ Wk  [2048 x 256]
    wmma_gemm<false, false, false, false>
        <<<dim3(DHEAD / 128, S_LEN / 128, 1), 256, 0, stream>>>(
        xb, 0, HID, wkbT, 0, HID, kb, 0, DHEAD, HID, 1.0f);
    // V^T = (X @ Wv)^T  [256 x 2048]  (epilogue writes transposed)
    wmma_gemm<false, false, true, false>
        <<<dim3(DHEAD / 128, S_LEN / 128, 1), 256, 0, stream>>>(
        xb, 0, HID, wvbT, 0, HID, vbT, 0, S_LEN, HID, 1.0f);

    // RoPE on Q (8 heads) and K
    rope_kernel<<<dim3(S_LEN), 128, 0, stream>>>(qb, kb, pos);

    // scores = scale * Q_h @ K^T + causal_mask -> fp32 attn region
    wmma_gemm<false, true, false, true>
        <<<dim3(S_LEN / 128, S_LEN / 128, NHEAD), 256, 0, stream>>>(
        qb, (long)DHEAD, HID,             // head offset = h*256 cols
        kb, 0, DHEAD,                     // B^T = K itself
        attnp, (long)S_LEN * S_LEN, S_LEN,
        DHEAD, kScale);

    // softmax in place -> this IS the attn output
    softmax_kernel<<<dim3(NHEAD * S_LEN), 256, 0, stream>>>(attnp);

    // ctx_h = attn_h @ V   (A fp32 staged->bf16; B^T = V^T)
    wmma_gemm<true, false, false, false>
        <<<dim3(DHEAD / 128, S_LEN / 128, NHEAD), 256, 0, stream>>>(
        attnp, (long)S_LEN * S_LEN, S_LEN,
        vbT, 0, S_LEN,
        ctx, (long)DHEAD, HID,            // head offset = h*256 cols
        S_LEN, 1.0f);

    // out = ctx @ Wo  -> fp32
    wmma_gemm<false, true, false, false>
        <<<dim3(HID / 128, S_LEN / 128, 1), 256, 0, stream>>>(
        ctx, 0, HID, wobT, 0, HID, outp, 0, HID, HID, 1.0f);
}
